// MoEBiEncoder_69810398429504
// MI455X (gfx1250) — matrix operand order repaired and checked by reading, not compiled, not run on hardware
//
#include <hip/hip_runtime.h>

typedef __attribute__((ext_vector_type(16))) _Float16 v16h;
typedef __attribute__((ext_vector_type(8)))  _Float16 v8h;
typedef __attribute__((ext_vector_type(8)))  float    v8f;

#define BATCH 8192
#define HDIM  1024
#define HHALF 512
#define NEXP  16

__device__ __forceinline__ float wave_reduce_sum(float v) {
#pragma unroll
  for (int off = 16; off > 0; off >>= 1) v += __shfl_xor(v, off, 32);
  return v;
}

// ---------------------------------------------------------------------------
// Elementwise fp32 -> f16 conversion (8 elems/thread, b128 in / b128 out).
// ---------------------------------------------------------------------------
__global__ __launch_bounds__(256) void f32_to_f16_kernel(
    const float* __restrict__ in, _Float16* __restrict__ out, int n)
{
  const int i = (blockIdx.x * 256 + threadIdx.x) * 8;
  if (i >= n) return;
  const float4 a = *(const float4*)(in + i);
  const float4 b = *(const float4*)(in + i + 4);
  v8h o;
  o[0] = (_Float16)a.x; o[1] = (_Float16)a.y;
  o[2] = (_Float16)a.z; o[3] = (_Float16)a.w;
  o[4] = (_Float16)b.x; o[5] = (_Float16)b.y;
  o[6] = (_Float16)b.z; o[7] = (_Float16)b.w;
  *(v8h*)(out + i) = o;
}

// ---------------------------------------------------------------------------
// Fused (optionally gathered) GEMM on pre-converted f16 operands:
//   Out[row, n] = act( sum_k A[row,k]*W[e][n,k] + bias[e][n] ) * (gate or 1)
// Wave strip: 32 rows x 64 cols = 2 A-frags x 4 B-frags -> 8 WMMAs / K-step.
// Block = 256 thr = 8 waves -> 32 x 512 tile. grid = (rows/32, Ntot/512, E).
// ---------------------------------------------------------------------------
__global__ __launch_bounds__(256) void fused_gemm_wmma(
    const _Float16* __restrict__ A, const _Float16* __restrict__ Wall,
    const float* __restrict__ bias_all, void* __restrict__ Out,
    const int* __restrict__ rowlist, const int* __restrict__ counts,
    const float* __restrict__ gatev,
    int K, int Ntot, int ldo, int relu_flag, int out_half)
{
  const int e      = blockIdx.z;
  const int n_rows = counts ? counts[e] : BATCH;
  const int tr     = blockIdx.x * 32;
  if (tr >= n_rows) return;

  const int  lane   = threadIdx.x & 31;
  const int  wv     = threadIdx.x >> 5;
  const int  lmod   = lane & 15;
  const bool hiHalf = lane >= 16;

  const _Float16* We    = Wall + (size_t)e * Ntot * K;
  const float*    be    = bias_all + (size_t)e * Ntot;
  const int       nbase = (blockIdx.y * 8 + wv) * 64;

  // Gather the A rows this lane supplies (M = lane % 16) for both row tiles.
  const _Float16* arow[2];
#pragma unroll
  for (int m = 0; m < 2; ++m) {
    const int slot  = tr + m * 16 + lmod;
    const int slotc = (slot < n_rows) ? slot : tr;        // clamp; masked later
    const int grow  = rowlist ? rowlist[(size_t)e * BATCH + slotc] : slotc;
    arow[m] = A + (size_t)grow * K;
  }

  v8f acc[2][4];
#pragma unroll
  for (int m = 0; m < 2; ++m)
#pragma unroll
    for (int t = 0; t < 4; ++t)
      acc[m][t] = (v8f){0.f,0.f,0.f,0.f,0.f,0.f,0.f,0.f};

  const int khiA = hiHalf ? 16 : 0;   // A-layout: hi lanes hold K+16..K+31
  const int khiB = hiHalf ? 16 : 0;   // B-layout: hi lanes hold K+16..K+31

  for (int k0 = 0; k0 < K; k0 += 32) {
    v16h af[2];
#pragma unroll
    for (int m = 0; m < 2; ++m) {
      const v8h lo = *(const v8h*)(arow[m] + k0 + khiA);        // K = +0..7
      const v8h hi = *(const v8h*)(arow[m] + k0 + khiA + 8);    // K = +8..15
      af[m] = __builtin_shufflevector(lo, hi, 0,1,2,3,4,5,6,7,
                                      8,9,10,11,12,13,14,15);
    }
#pragma unroll
    for (int t = 0; t < 4; ++t) {
      const v16h bf =
          *(const v16h*)(We + (size_t)(nbase + t * 16 + lmod) * K + k0 + khiB);
      acc[0][t] = __builtin_amdgcn_wmma_f32_16x16x32_f16(
          false, af[0], false, bf, (short)0, acc[0][t], false, false);
      acc[1][t] = __builtin_amdgcn_wmma_f32_16x16x32_f16(
          false, af[1], false, bf, (short)0, acc[1][t], false, false);
    }
  }

  // ---- Epilogue: bias, activation, optional gate, scatter store ----
#pragma unroll
  for (int m = 0; m < 2; ++m) {
#pragma unroll
    for (int t = 0; t < 4; ++t) {
      const int   ncol = nbase + t * 16 + lmod;     // D: N = lane % 16
      const float bv   = be[ncol];
#pragma unroll
      for (int r = 0; r < 8; ++r) {
        const int slot = tr + m * 16 + r + (hiHalf ? 8 : 0);  // D: M = r+8*hi
        if (slot < n_rows) {
          float v = acc[m][t][r] + bv;
          if (relu_flag) v = fmaxf(v, 0.f);
          const int grow = rowlist ? rowlist[(size_t)e * BATCH + slot] : slot;
          if (gatev) v *= gatev[(size_t)e * BATCH + slot];
          const size_t idx = (size_t)grow * ldo + ncol;
          if (out_half) ((_Float16*)Out)[idx] = (_Float16)v;
          else          ((float*)Out)[idx]    = v;
        }
      }
    }
  }
}

// ---------------------------------------------------------------------------
// Router: wave per row. logits = x1 @ cls3_w^T + b; softmax -> top1 gate;
// build per-expert buckets with atomics.
// ---------------------------------------------------------------------------
__global__ __launch_bounds__(256) void router_kernel(
    const float* __restrict__ x1, const float* __restrict__ w3,
    const float* __restrict__ b3, int* __restrict__ counts,
    int* __restrict__ rowlist, float* __restrict__ gates)
{
  const int wv   = threadIdx.x >> 5;
  const int lane = threadIdx.x & 31;
  const int row  = blockIdx.x * 8 + wv;
  if (row >= BATCH) return;

  const float* xr = x1 + (size_t)row * HHALF + lane * 16;
  float xv[16];
#pragma unroll
  for (int i = 0; i < 16; ++i) xv[i] = xr[i];

  float logit[NEXP];
#pragma unroll
  for (int m = 0; m < NEXP; ++m) {
    const float* wr = w3 + (size_t)m * HHALF + lane * 16;
    float p = 0.f;
#pragma unroll
    for (int i = 0; i < 16; ++i) p += xv[i] * wr[i];
    logit[m] = wave_reduce_sum(p) + b3[m];   // every lane ends with full sum
  }

  if (lane == 0) {
    float mx = logit[0]; int am = 0;
#pragma unroll
    for (int m = 1; m < NEXP; ++m)
      if (logit[m] > mx) { mx = logit[m]; am = m; }   // first-max, like argmax
    float s = 0.f;
#pragma unroll
    for (int m = 0; m < NEXP; ++m) s += expf(logit[m] - mx);
    const float gate = 1.0f / s;          // top-1 softmax prob
    const int pos = atomicAdd(&counts[am], 1);
    rowlist[(size_t)am * BATCH + pos] = row;
    gates  [(size_t)am * BATCH + pos] = gate;
  }
}

// ---------------------------------------------------------------------------
// out[row] = out[row] / max(||out[row]||2, 1e-6) + q[row]  (in place, wave/row)
// ---------------------------------------------------------------------------
__global__ __launch_bounds__(256) void norm_add_kernel(
    float* __restrict__ out, const float* __restrict__ q)
{
  const int wv   = threadIdx.x >> 5;
  const int lane = threadIdx.x & 31;
  const int row  = blockIdx.x * 8 + wv;
  if (row >= BATCH) return;

  float*       orow = out + (size_t)row * HDIM;
  const float* qrow = q   + (size_t)row * HDIM;

  float vals[32];
  float ss = 0.f;
#pragma unroll
  for (int i = 0; i < 32; ++i) {
    const float v = orow[lane + i * 32];
    vals[i] = v;
    ss += v * v;
  }
  ss = wave_reduce_sum(ss);
  const float inv = 1.0f / fmaxf(sqrtf(ss), 1e-6f);
#pragma unroll
  for (int i = 0; i < 32; ++i)
    orow[lane + i * 32] = vals[i] * inv + qrow[lane + i * 32];
}

// ---------------------------------------------------------------------------
extern "C" void kernel_launch(void* const* d_in, const int* in_sizes, int n_in,
                              void* d_out, int out_size, void* d_ws, size_t ws_size,
                              hipStream_t stream) {
  const float* q      = (const float*)d_in[0];  // [B, H]
  const float* cls1_w = (const float*)d_in[1];  // [Hh, H]
  const float* cls1_b = (const float*)d_in[2];  // [Hh]
  const float* cls3_w = (const float*)d_in[3];  // [M, Hh]
  const float* cls3_b = (const float*)d_in[4];  // [M]
  const float* exp_w1 = (const float*)d_in[5];  // [M, Hh, H]
  const float* exp_b1 = (const float*)d_in[6];  // [M, Hh]
  const float* exp_w2 = (const float*)d_in[7];  // [M, H, Hh]
  const float* exp_b2 = (const float*)d_in[8];  // [M, H]
  float*       out    = (float*)d_out;          // [B, H]

  char* ws = (char*)d_ws;
  size_t off = 0;
  _Float16* qh   = (_Float16*)(ws + off); off += (size_t)BATCH * HDIM  * 2;
  _Float16* c1h  = (_Float16*)(ws + off); off += (size_t)HHALF * HDIM  * 2;
  _Float16* w1h  = (_Float16*)(ws + off); off += (size_t)NEXP * HHALF * HDIM * 2;
  _Float16* w2h  = (_Float16*)(ws + off); off += (size_t)NEXP * HDIM * HHALF * 2;
  _Float16* hbuf = (_Float16*)(ws + off); off += (size_t)BATCH * HHALF * 2;
  float*    x1   = (float*)   (ws + off); off += (size_t)BATCH * HHALF * 4;
  int*    counts  = (int*)  (ws + off); off += 256;
  int*    rowlist = (int*)  (ws + off); off += (size_t)NEXP * BATCH * 4;
  float*  gates   = (float*)(ws + off); off += (size_t)NEXP * BATCH * 4;

  hipMemsetAsync(counts, 0, NEXP * sizeof(int), stream);

  const dim3 blk(256, 1, 1);

  // 0) One-shot fp32 -> f16 conversions (memory-bound, ~5us total)
  {
    const int nq  = BATCH * HDIM;                 // 8M
    const int nc1 = HHALF * HDIM;                 // 0.5M
    const int nw  = NEXP * HHALF * HDIM;          // 8M each
    f32_to_f16_kernel<<<dim3(nq  / (256*8)), blk, 0, stream>>>(q,      qh,  nq);
    f32_to_f16_kernel<<<dim3(nc1 / (256*8)), blk, 0, stream>>>(cls1_w, c1h, nc1);
    f32_to_f16_kernel<<<dim3(nw  / (256*8)), blk, 0, stream>>>(exp_w1, w1h, nw);
    f32_to_f16_kernel<<<dim3(nw  / (256*8)), blk, 0, stream>>>(exp_w2, w2h, nw);
  }

  // 1) Gating GEMM: x1 = relu(q @ cls1_w^T + cls1_b)   [B, Hh] fp32
  fused_gemm_wmma<<<dim3(BATCH / 32, 1, 1), blk, 0, stream>>>(
      qh, c1h, cls1_b, x1, nullptr, nullptr, nullptr,
      HDIM, HHALF, HHALF, /*relu=*/1, /*out_half=*/0);

  // 2) Router: logits/softmax/top1 -> buckets
  router_kernel<<<dim3(BATCH / 8, 1, 1), blk, 0, stream>>>(
      x1, cls3_w, cls3_b, counts, rowlist, gates);

  // 3) Expert layer 1 (gathered): h = relu(q_rows @ w1[e]^T + b1[e]) -> f16
  fused_gemm_wmma<<<dim3(BATCH / 32, 1, NEXP), blk, 0, stream>>>(
      qh, w1h, exp_b1, hbuf, rowlist, counts, nullptr,
      HDIM, HHALF, HHALF, /*relu=*/1, /*out_half=*/1);

  // 4) Expert layer 2 (gathered, gate-scaled, scattered into d_out) fp32
  fused_gemm_wmma<<<dim3(BATCH / 32, 2, NEXP), blk, 0, stream>>>(
      hbuf, w2h, exp_b2, out, rowlist, counts, gates,
      HHALF, HDIM, HDIM, /*relu=*/0, /*out_half=*/0);

  // 5) out = l2_normalize(out) + q
  norm_add_kernel<<<dim3(BATCH / 8, 1, 1), blk, 0, stream>>>(out, q);
}
